// DCT2_83605833384446
// MI455X (gfx1250) — compile-verified, matrix operand order checked by reading, not run on hardware
//
#include <hip/hip_runtime.h>

// ---------------------------------------------------------------------------
// 2D DCT-II of a 4096x4096 fp32 matrix on MI455X (gfx1250, wave32).
//   out = C * x * C^T,  C[k,i] = cos(pi*(2i+1)*k / 8192)
// fp32-accurate GEMMs on the f16 matrix pipes via the 2-term split trick:
//   A*B ~= Ahi*Bhi + Ahi*Blo + Alo*Bhi   (f16 operands, f32 accumulate)
// using V_WMMA_F32_16X16X32_F16, with double-buffered LDS fed by the CDNA5
// async global->LDS path (GLOBAL_LOAD_ASYNC_TO_LDS_B128 / ASYNCcnt).
// ---------------------------------------------------------------------------

typedef _Float16 f16;
typedef __attribute__((ext_vector_type(8)))  _Float16 f16x8;
typedef __attribute__((ext_vector_type(16))) _Float16 v16h;
typedef __attribute__((ext_vector_type(8)))  float    v8f;
typedef int b128_t __attribute__((vector_size(16)));   // 16-byte chunk type

#define DCT_N 4096
#define BM 128
#define BN 128
#define BK 64
#define LDH 72            // LDS row stride in f16 (64 + 8 pad; 144B, 16B-aligned)

#if defined(__gfx1250__) && __has_builtin(__builtin_amdgcn_global_load_async_to_lds_b128)
#define USE_ASYNC 1
#else
#define USE_ASYNC 0
#endif

// 16-byte global->LDS copy: async DMA-style on CDNA5, sync fallback otherwise.
static __device__ __forceinline__ void copy16_g2l(const f16* g, f16* l) {
#if USE_ASYNC
  __builtin_amdgcn_global_load_async_to_lds_b128(
      (__attribute__((address_space(1))) b128_t*)g,
      (__attribute__((address_space(3))) b128_t*)l, 0, 0);
#else
  *(f16x8*)l = *(const f16x8*)g;
#endif
}

static __device__ __forceinline__ void wait_async_copies() {
#if USE_ASYNC
#if __has_builtin(__builtin_amdgcn_s_wait_asynccnt)
  __builtin_amdgcn_s_wait_asynccnt(0);
#else
  asm volatile("s_wait_asynccnt 0x0" ::: "memory");
#endif
#endif
}

// ---------------------------------------------------------------------------
// DCT basis, split into f16 hi/lo. Integer phase mod 4N (= 2^14) keeps the
// cosf argument in [0, 2*pi) exactly -> no range-reduction error.
// ---------------------------------------------------------------------------
__global__ void dct_gen_cos_f16_kernel(f16* __restrict__ Chi,
                                       f16* __restrict__ Clo) {
  const int idx = blockIdx.x * blockDim.x + threadIdx.x;   // < 4096*4096
  const int k = idx >> 12;
  const int i = idx & (DCT_N - 1);
  const unsigned phase = ((unsigned)(2 * i + 1) * (unsigned)k) & (4u * DCT_N - 1u);
  const float c = cosf((float)phase * (float)(3.14159265358979323846 / (2.0 * DCT_N)));
  const f16 h = (f16)c;
  Chi[idx] = h;
  Clo[idx] = (f16)(c - (float)h);
}

// ---------------------------------------------------------------------------
// Transpose-split x:  xT_hi/lo[n][k] = split(x[k][n]).  64x64 LDS tiles.
// ---------------------------------------------------------------------------
__global__ __launch_bounds__(256)
void dct_split_transpose_kernel(const float* __restrict__ X,
                                f16* __restrict__ Thi,
                                f16* __restrict__ Tlo) {
  __shared__ float tile[64][65];
  const int tx = threadIdx.x & 63;
  const int ty = threadIdx.x >> 6;          // 0..3
  const int r0 = blockIdx.y * 64;           // k block
  const int c0 = blockIdx.x * 64;           // n block
#pragma unroll
  for (int i = 0; i < 16; ++i) {
    const int r = ty + i * 4;
    tile[r][tx] = X[(size_t)(r0 + r) * DCT_N + c0 + tx];
  }
  __syncthreads();
#pragma unroll
  for (int i = 0; i < 16; ++i) {
    const int r2 = ty + i * 4;              // local n
    const float v = tile[tx][r2];           // x[r0+tx][c0+r2]
    const f16 h = (f16)v;
    const size_t o = (size_t)(c0 + r2) * DCT_N + r0 + tx;
    Thi[o] = h;
    Tlo[o] = (f16)(v - (float)h);
  }
}

// ---------------------------------------------------------------------------
// Fragment assembly from K-minor LDS. 16-bit A/B operand layout (ISA 7.12.2):
// lanes 0-15 hold K {base..base+7, base+16..base+23}; lanes 16-31 hold
// K {base+8..base+15, base+24..base+31}. Two aligned 16B loads + shuffle.
// ---------------------------------------------------------------------------
static __device__ __forceinline__ v16h load_frag(const f16* row, int base) {
  const f16x8 p0 = *(const f16x8*)(row + base);
  const f16x8 p1 = *(const f16x8*)(row + base + 16);
  return __builtin_shufflevector(p0, p1, 0, 1, 2, 3, 4, 5, 6, 7,
                                 8, 9, 10, 11, 12, 13, 14, 15);
}

static __device__ __forceinline__ v8f wmma16(v16h a, v16h b, v8f c) {
  return __builtin_amdgcn_wmma_f32_16x16x32_f16(false, a, false, b,
                                                (short)0, c, false, false);
}

// ---------------------------------------------------------------------------
// f16x3 GEMM with double-buffered async-staged LDS.
//   D[M,N] = (Ahi+Alo)[M,K] * B, where B arrays are stored [n][k] (K-minor).
// 256 threads = 8 waves (2M x 4N); each wave owns 64x32 = 4x2 WMMA tiles,
// 3 wmma per tile per 32-wide k-step. SPLIT_OUT: emit f16 hi/lo (feeds
// pass 2) instead of f32.
// ---------------------------------------------------------------------------
template <bool SPLIT_OUT>
__global__ __launch_bounds__(256)
void dct_gemm_f16x3_kernel(const f16* __restrict__ Ahi, const f16* __restrict__ Alo,
                           const f16* __restrict__ Bhi, const f16* __restrict__ Blo,
                           float* __restrict__ Df32,
                           f16* __restrict__ Dhi, f16* __restrict__ Dlo) {
  __shared__ f16 sAhi[2][BM][LDH];
  __shared__ f16 sAlo[2][BM][LDH];
  __shared__ f16 sBhi[2][BN][LDH];
  __shared__ f16 sBlo[2][BN][LDH];   // total 144 KB

  const int t     = threadIdx.x;
  const int lane  = t & 31;
  const int wave  = t >> 5;
  const int waveM = wave & 1;    // 0..1 -> 64-row slab
  const int waveN = wave >> 1;   // 0..3 -> 32-col slab
  const int lh    = lane & 15;
  const int half  = lane >> 4;

  const int blockM = blockIdx.y * BM;
  const int blockN = blockIdx.x * BN;

  auto stage = [&](int buf, int k0) {
#pragma unroll
    for (int i = 0; i < 4; ++i) {
      const int idx = t + i * 256;       // 0..1023 chunk slots
      const int r   = idx >> 3;          // 8 chunks per 64-f16 row
      const int c8  = (idx & 7) << 3;
      const size_t ga = (size_t)(blockM + r) * DCT_N + k0 + c8;
      const size_t gb = (size_t)(blockN + r) * DCT_N + k0 + c8;
      copy16_g2l(Ahi + ga, &sAhi[buf][r][c8]);
      copy16_g2l(Alo + ga, &sAlo[buf][r][c8]);
      copy16_g2l(Bhi + gb, &sBhi[buf][r][c8]);
      copy16_g2l(Blo + gb, &sBlo[buf][r][c8]);
    }
  };

  const v8f vzero = {0.f, 0.f, 0.f, 0.f, 0.f, 0.f, 0.f, 0.f};
  v8f acc[4][2];
#pragma unroll
  for (int wm = 0; wm < 4; ++wm)
#pragma unroll
    for (int wn = 0; wn < 2; ++wn) acc[wm][wn] = vzero;

  // Prologue: fill buffer 0.
  stage(0, 0);
  wait_async_copies();
  __syncthreads();

  int buf = 0;
  for (int k0 = 0; k0 < DCT_N; k0 += BK) {
    // Prefetch next K-tile into the other buffer while we compute.
    if (k0 + BK < DCT_N) stage(buf ^ 1, k0 + BK);

#pragma unroll
    for (int ks = 0; ks < BK; ks += 32) {
      const int base = ks + (half << 3);
      v16h bhi[2], blo[2];
#pragma unroll
      for (int wn = 0; wn < 2; ++wn) {
        const int nrow = waveN * 32 + wn * 16 + lh;
        bhi[wn] = load_frag(&sBhi[buf][nrow][0], base);
        blo[wn] = load_frag(&sBlo[buf][nrow][0], base);
      }
#pragma unroll
      for (int wm = 0; wm < 4; ++wm) {
        const int mrow = waveM * 64 + wm * 16 + lh;
        const v16h ahi = load_frag(&sAhi[buf][mrow][0], base);
        const v16h alo = load_frag(&sAlo[buf][mrow][0], base);
        // interleave over wn so consecutive wmma hit different accumulators
#pragma unroll
        for (int wn = 0; wn < 2; ++wn) acc[wm][wn] = wmma16(ahi, bhi[wn], acc[wm][wn]);
#pragma unroll
        for (int wn = 0; wn < 2; ++wn) acc[wm][wn] = wmma16(ahi, blo[wn], acc[wm][wn]);
#pragma unroll
        for (int wn = 0; wn < 2; ++wn) acc[wm][wn] = wmma16(alo, bhi[wn], acc[wm][wn]);
      }
    }

    // My prefetch must have landed; then all waves agree before buffer swap.
    wait_async_copies();
    __syncthreads();
    buf ^= 1;
  }

  // ---- epilogue: C/D layout -> lanes 0-15: M=r, lanes 16-31: M=r+8 ----
#pragma unroll
  for (int wm = 0; wm < 4; ++wm) {
#pragma unroll
    for (int wn = 0; wn < 2; ++wn) {
      const int rbase = blockM + waveM * 64 + wm * 16 + half * 8;
      const int cbase = blockN + waveN * 32 + wn * 16 + lh;
#pragma unroll
      for (int r = 0; r < 8; ++r) {
        const float v = acc[wm][wn][r];
        const size_t o = (size_t)(rbase + r) * DCT_N + cbase;
        if (SPLIT_OUT) {
          const f16 h = (f16)v;
          Dhi[o] = h;
          Dlo[o] = (f16)(v - (float)h);
        } else {
          Df32[o] = v;
        }
      }
    }
  }
}

// ---------------------------------------------------------------------------
extern "C" void kernel_launch(void* const* d_in, const int* in_sizes, int n_in,
                              void* d_out, int out_size, void* d_ws,
                              size_t ws_size, hipStream_t stream) {
  (void)in_sizes; (void)n_in; (void)out_size; (void)ws_size;

  const float* x = (const float*)d_in[0];
  float* out = (float*)d_out;

  const size_t NN = (size_t)DCT_N * DCT_N;
  f16* Chi = (f16*)d_ws;          // 32 MB each; 6 buffers = 192 MB total
  f16* Clo = Chi + NN;
  f16* Xhi = Clo + NN;            // transposed x, split
  f16* Xlo = Xhi + NN;
  f16* Thi = Xlo + NN;            // tmp = C*x, split
  f16* Tlo = Thi + NN;

  // 1) DCT basis (hi/lo f16)
  dct_gen_cos_f16_kernel<<<(DCT_N * DCT_N) / 256, 256, 0, stream>>>(Chi, Clo);

  // 2) xT (hi/lo f16), K-minor for pass-1 B operand
  dct_split_transpose_kernel<<<dim3(64, 64), 256, 0, stream>>>(x, Xhi, Xlo);

  // 3) tmp = C * x   (A = C row-major, B = xT [n][k]); split epilogue
  dim3 grid(DCT_N / BN, DCT_N / BM);
  dct_gemm_f16x3_kernel<true><<<grid, 256, 0, stream>>>(
      Chi, Clo, Xhi, Xlo, nullptr, Thi, Tlo);

  // 4) out = tmp * C^T (A = tmp row-major, B = C rows as [n][k]); f32 out
  dct_gemm_f16x3_kernel<false><<<grid, 256, 0, stream>>>(
      Thi, Tlo, Chi, Clo, out, nullptr, nullptr);
}
